// ECPAttention_68736656605501
// MI455X (gfx1250) — compile-verified, matrix operand order checked by reading, not run on hardware
//
#include <hip/hip_runtime.h>
#include <hip/hip_bf16.h>

typedef __attribute__((ext_vector_type(4)))  __bf16 v4bf;
typedef __attribute__((ext_vector_type(8)))  __bf16 v8bf;
typedef __attribute__((ext_vector_type(16))) __bf16 v16bf;
typedef __attribute__((ext_vector_type(8)))  float  v8f;
typedef __attribute__((ext_vector_type(4)))  unsigned int u32x4;
typedef __attribute__((ext_vector_type(8)))  int    i32x8;
typedef __attribute__((ext_vector_type(4)))  int    i32x4;

#define HEADS 16
#define DHEAD 64
#define SEQN  4096
#define DIM   1024
#define SEGM  512

__device__ __forceinline__ v8f wmma_bf16(v16bf a, v16bf b, v8f c) {
  // 8 args: (neg_a, A, neg_b, B, c_mod, C, reuse_a, reuse_b)
  return __builtin_amdgcn_wmma_f32_16x16x32_bf16(false, a, false, b, (short)0, c, false, false);
}

__device__ __forceinline__ v16bf frag16(const __bf16* p0, const __bf16* p1) {
  // A/B fragment: elements 0..7 from p0 (k = h*8..), 8..15 from p1 (k = 16+h*8..)
  v8bf lo = *(const v8bf*)p0;
  v8bf hi = *(const v8bf*)p1;
  v16bf r;
#pragma unroll
  for (int i = 0; i < 8; ++i) { r[i] = lo[i]; r[i + 8] = hi[i]; }
  return r;
}

__device__ __forceinline__ v8f vzero8() {
  v8f z;
#pragma unroll
  for (int i = 0; i < 8; ++i) z[i] = 0.0f;
  return z;
}

// ---------------------------------------------------------------------------
// Kernel 0: RoPE cos/sin tables, one entry per (pos, dd): 4096 x 64 each.
// inv_freq[i] = 10000^(-i/32) = exp2(-i * log2(10000)/32)
// ---------------------------------------------------------------------------
__global__ __launch_bounds__(256)
void rope_table_kernel(float* __restrict__ cosT, float* __restrict__ sinT) {
  int idx = blockIdx.x * 256 + threadIdx.x;   // 0 .. 4096*64-1
  int pos = idx >> 6, dd = idx & 63;
  float ang = (float)pos * exp2f((float)(dd & 31) * -0.41524101186092437f);
  cosT[idx] = __cosf(ang);
  sinT[idx] = __sinf(ang);
}

// ---------------------------------------------------------------------------
// Kernel 1: Y = RoPE(X @ W) [* d^-0.5 if is_q], bf16 head-major [b*h][pos][dd]
// Block: 256 threads (8 waves), tile = 128 rows x 64 cols, K-steps of 32.
// ---------------------------------------------------------------------------
__global__ __launch_bounds__(256)
void proj_rope_kernel(const float* __restrict__ X, const float* __restrict__ W,
                      const float* __restrict__ cosT,
                      const float* __restrict__ sinT,
                      __bf16* __restrict__ Y, int is_q) {
  __shared__ __bf16 Xs[128 * 40];  // 128 rows x 32 k (stride 40: bank pad)
  __shared__ __bf16 Wst[64 * 40];  // W transposed: 64 cols x 32 k

  const int tid = threadIdx.x;
  const int w = tid >> 5, lane = tid & 31, l16 = lane & 15, h = lane >> 4;
  const int colbase = blockIdx.x * 64;
  const int rowbase = blockIdx.y * 128;

  v8f acc[4];
#pragma unroll
  for (int t = 0; t < 4; ++t) acc[t] = vzero8();

  for (int kb = 0; kb < DIM; kb += 32) {
    // X tile: 128x32 fp32 as 1024 float4 loads -> bf16 ds_store_b64.
    // Unrolled (4 per thread) so all global loads issue before the waits.
#pragma unroll
    for (int k4 = 0; k4 < 4; ++k4) {
      int idx = tid + k4 * 256;
      int i = idx >> 3, j4 = (idx & 7) * 4;
      float4 v = *(const float4*)&X[(size_t)(rowbase + i) * DIM + kb + j4];
      v4bf b;
      b[0] = (__bf16)v.x; b[1] = (__bf16)v.y; b[2] = (__bf16)v.z; b[3] = (__bf16)v.w;
      *(v4bf*)&Xs[i * 40 + j4] = b;
    }
    // W tile: 32x64 fp32 as 512 float4 loads -> transposed bf16 stores
#pragma unroll
    for (int k4 = 0; k4 < 2; ++k4) {
      int idx = tid + k4 * 256;
      int k = idx >> 4, c4 = (idx & 15) * 4;
      float4 v = *(const float4*)&W[(size_t)(kb + k) * DIM + colbase + c4];
      Wst[(c4 + 0) * 40 + k] = (__bf16)v.x;
      Wst[(c4 + 1) * 40 + k] = (__bf16)v.y;
      Wst[(c4 + 2) * 40 + k] = (__bf16)v.z;
      Wst[(c4 + 3) * 40 + k] = (__bf16)v.w;
    }
    __syncthreads();

    const int arow = w * 16 + l16;
    v16bf a = frag16(&Xs[arow * 40 + h * 8], &Xs[arow * 40 + 16 + h * 8]);
#pragma unroll
    for (int t = 0; t < 4; ++t) {
      int bcol = t * 16 + l16;
      v16bf b = frag16(&Wst[bcol * 40 + h * 8], &Wst[bcol * 40 + 16 + h * 8]);
      acc[t] = wmma_bf16(a, b, acc[t]);
    }
    __syncthreads();
  }

  // RoPE epilogue: column dd pairs with dd^32, i.e. accumulator t <-> t^2.
  const int head = colbase >> 6;
#pragma unroll
  for (int r = 0; r < 8; ++r) {
    int grow = rowbase + w * 16 + r + h * 8;     // global row (b*n)
    int pos = grow & (SEQN - 1);
    int headg = (grow >> 12) * HEADS + head;     // b_idx*16 + head
#pragma unroll
    for (int t = 0; t < 4; ++t) {
      int dd = t * 16 + l16;
      float cv = cosT[pos * DHEAD + dd];
      float sv = sinT[pos * DHEAD + dd];
      float vv = acc[t][r];
      float ov = acc[t ^ 2][r];
      float outv = (dd < 32) ? (vv * cv - ov * sv) : (vv * cv + ov * sv);
      if (is_q) outv *= 0.125f;  // d^-0.5, d=64
      Y[((size_t)headg * SEQN + pos) * DHEAD + dd] = (__bf16)outv;
    }
  }
}

// ---------------------------------------------------------------------------
// Kernel 2: segmented flash attention.
// Block = (query-tile qt of 128, segment, channel). 8 waves x 16 query rows.
// K tile staged by the Tensor Data Mover (TDM); V tile staged transposed.
// ---------------------------------------------------------------------------
__global__ __launch_bounds__(256)
void attn_kernel(const __bf16* __restrict__ Qb, const __bf16* __restrict__ KVb,
                 __bf16* __restrict__ A2) {
  __shared__ __bf16 Ks[32 * 72];      // 32 keys x 64 d (+16B row pad via TDM)
  __shared__ __bf16 Vs[64 * 40];      // transposed: 64 d x 32 keys
  __shared__ __bf16 Ps[8 * 16 * 40];  // per-wave P tile: 16 q x 32 keys

  const int tid = threadIdx.x;
  const int w = tid >> 5, lane = tid & 31, l16 = lane & 15, h = lane >> 4;
  const int qt = blockIdx.x, seg = blockIdx.y, ch = blockIdx.z;

  const int q0 = qt * 128;
  const int kvstart = (seg == 0) ? 0 : (seg - 1) * SEGM;
  const int causal_off = (seg == 0) ? 0 : SEGM;
  const int jmax = causal_off + q0 + 128;  // skip fully-masked causal blocks

  // Q fragments (two K-chunks of 32 over d=64), loaded once from global.
  const __bf16* qp =
      Qb + ((size_t)ch * SEQN + seg * SEGM + q0 + w * 16 + l16) * DHEAD;
  v16bf qa0 = frag16(qp + h * 8, qp + 16 + h * 8);
  v16bf qa1 = frag16(qp + 32 + h * 8, qp + 48 + h * 8);

  float mrow[8], lrow[8];
  v8f acc[4];
#pragma unroll
  for (int r = 0; r < 8; ++r) { mrow[r] = -1e30f; lrow[r] = 0.0f; }
#pragma unroll
  for (int t = 0; t < 4; ++t) acc[t] = vzero8();

  __bf16* Pw = &Ps[w * 16 * 40];
  const unsigned int ksLds = (unsigned int)(size_t)&Ks[0];

  for (int jb = 0; jb < jmax; jb += 32) {
    const __bf16* kvTile =
        KVb + ((size_t)ch * SEQN + kvstart + jb) * DHEAD;

    if (w == 0) {
      // --- TDM descriptor: 2D tile 64 x 32 of 2-byte elems, padded rows ---
      unsigned long long ga = (unsigned long long)(size_t)kvTile;
      u32x4 g0;
      g0[0] = 1u;                                       // count=1, load D#
      g0[1] = ksLds;                                    // lds_addr
      g0[2] = (unsigned int)(ga & 0xffffffffu);         // global_addr[31:0]
      g0[3] = (unsigned int)((ga >> 32) & 0x1ffffffu)   // global_addr[56:32]
              | (2u << 30);                             // type=2 (image)
      i32x8 g1;
      g1[0] = (1 << 16)      // data_size = 2 bytes
            | (1 << 20)      // pad_enable (LDS row padding)
            | (4 << 22)      // pad_interval: every 32 DWORDs (=128B row)
            | (3 << 25);     // pad_amount: 4 DWORDs (=16B -> stride 72 elems)
      g1[1] = (DHEAD & 0xffff) << 16;                   // tensor_dim0 = 64
      g1[2] = (SEQN & 0xffff) << 16;                    // tensor_dim1 = 4096
      g1[3] = (DHEAD & 0xffff) << 16;                   // tile_dim0 = 64
      g1[4] = 32;                                       // tile_dim1 = 32 keys
      g1[5] = DHEAD;                                    // tensor_dim0_stride
      g1[6] = 0;
      g1[7] = 0;
      i32x4 gz4;
      gz4[0] = 0; gz4[1] = 0; gz4[2] = 0; gz4[3] = 0;
      i32x8 gz8;
#pragma unroll
      for (int i = 0; i < 8; ++i) gz8[i] = 0;
      // amdgpu-toolchain (clang-23) form: 6 args, trailing i32x8 + cpol.
      __builtin_amdgcn_tensor_load_to_lds(g0, g1, gz4, gz4, gz8, 0);
    }

    // V tile transposed, vectorized: one b128 global load + 8 b16 LDS stores.
    {
      int key = tid >> 3, ddc = (tid & 7) * 8;
      v8bf v = *(const v8bf*)&kvTile[(size_t)key * DHEAD + ddc];
#pragma unroll
      for (int i = 0; i < 8; ++i) Vs[(ddc + i) * 40 + key] = v[i];
    }

    if (w == 0) __builtin_amdgcn_s_wait_tensorcnt(0);
    __syncthreads();

    // S (16 q x 32 keys) = Q · K^T
    v8f s[2];
#pragma unroll
    for (int t = 0; t < 2; ++t) {
      int key = t * 16 + l16;
      v16bf b0 = frag16(&Ks[key * 72 + h * 8], &Ks[key * 72 + 16 + h * 8]);
      s[t] = wmma_bf16(qa0, b0, vzero8());
      v16bf b1 = frag16(&Ks[key * 72 + 32 + h * 8], &Ks[key * 72 + 48 + h * 8]);
      s[t] = wmma_bf16(qa1, b1, s[t]);
    }

    // Online softmax. C-layout: row = r + h*8 lives entirely in one 16-lane
    // half, col = l16 + 16t -> width-16 xor reductions give full-row stats.
#pragma unroll
    for (int r = 0; r < 8; ++r) {
      int qg = q0 + w * 16 + r + h * 8;  // query index within segment
      float v0 = s[0][r], v1 = s[1][r];
      int j0 = jb + l16, j1 = jb + 16 + l16;
      if (j0 >= causal_off && (j0 - causal_off) > qg) v0 = -1e30f;
      if (j1 >= causal_off && (j1 - causal_off) > qg) v1 = -1e30f;
      float mx = fmaxf(v0, v1);
#pragma unroll
      for (int off = 8; off >= 1; off >>= 1)
        mx = fmaxf(mx, __shfl_xor(mx, off, 16));
      float mnew = fmaxf(mrow[r], mx);
      float sc = __expf(mrow[r] - mnew);
      mrow[r] = mnew;
      float p0 = (v0 < -1e29f) ? 0.0f : __expf(v0 - mnew);
      float p1 = (v1 < -1e29f) ? 0.0f : __expf(v1 - mnew);
      float rs = p0 + p1;
#pragma unroll
      for (int off = 8; off >= 1; off >>= 1) rs += __shfl_xor(rs, off, 16);
      lrow[r] = lrow[r] * sc + rs;
#pragma unroll
      for (int t = 0; t < 4; ++t) acc[t][r] *= sc;
      int prow = r + h * 8;
      Pw[prow * 40 + l16] = (__bf16)p0;        // C-layout -> LDS row-major
      Pw[prow * 40 + 16 + l16] = (__bf16)p1;
    }

    // O (16 q x 64 d) += P · V ; P re-read from LDS in A-fragment layout.
    v16bf pa = frag16(&Pw[l16 * 40 + h * 8], &Pw[l16 * 40 + 16 + h * 8]);
#pragma unroll
    for (int t = 0; t < 4; ++t) {
      int dcol = t * 16 + l16;
      v16bf vb = frag16(&Vs[dcol * 40 + h * 8], &Vs[dcol * 40 + 16 + h * 8]);
      acc[t] = wmma_bf16(pa, vb, acc[t]);
    }
    __syncthreads();
  }

  // Normalize and write merged-head bf16 activation [b][pos][h*64+dd].
  const int bidx = ch >> 4, hh = ch & 15;
#pragma unroll
  for (int r = 0; r < 8; ++r) {
    float inv = 1.0f / lrow[r];
    int pos = seg * SEGM + q0 + w * 16 + r + h * 8;
#pragma unroll
    for (int t = 0; t < 4; ++t) {
      int col = hh * 64 + t * 16 + l16;
      A2[((size_t)bidx * SEQN + pos) * DIM + col] = (__bf16)(acc[t][r] * inv);
    }
  }
}

// ---------------------------------------------------------------------------
// Kernel 3: out = A2 @ (Wo0|Wo) + (bo0|bo), weight chosen per row region.
// ---------------------------------------------------------------------------
__global__ __launch_bounds__(256)
void out_proj_kernel(const __bf16* __restrict__ A2,
                     const float* __restrict__ Wo, const float* __restrict__ bo,
                     const float* __restrict__ Wo0, const float* __restrict__ bo0,
                     float* __restrict__ out) {
  __shared__ __bf16 As[128 * 40];
  __shared__ __bf16 Wst[64 * 40];

  const int tid = threadIdx.x;
  const int w = tid >> 5, lane = tid & 31, l16 = lane & 15, h = lane >> 4;
  const int colbase = blockIdx.x * 64;
  const int rowbase = blockIdx.y * 128;
  const bool first_seg = ((rowbase & (SEQN - 1)) < SEGM);  // 512 tile-aligned
  const float* W = first_seg ? Wo0 : Wo;
  const float* bias = first_seg ? bo0 : bo;

  v8f acc[4];
#pragma unroll
  for (int t = 0; t < 4; ++t) acc[t] = vzero8();

  for (int kb = 0; kb < DIM; kb += 32) {
    // A tile already bf16: 512 x b128 copies (8 bf16 each), unrolled.
#pragma unroll
    for (int k4 = 0; k4 < 2; ++k4) {
      int idx = tid + k4 * 256;
      int i = idx >> 2, j8 = (idx & 3) * 8;
      *(v8bf*)&As[i * 40 + j8] =
          *(const v8bf*)&A2[(size_t)(rowbase + i) * DIM + kb + j8];
    }
#pragma unroll
    for (int k4 = 0; k4 < 2; ++k4) {
      int idx = tid + k4 * 256;
      int k = idx >> 4, c4 = (idx & 15) * 4;
      float4 v = *(const float4*)&W[(size_t)(kb + k) * DIM + colbase + c4];
      Wst[(c4 + 0) * 40 + k] = (__bf16)v.x;
      Wst[(c4 + 1) * 40 + k] = (__bf16)v.y;
      Wst[(c4 + 2) * 40 + k] = (__bf16)v.z;
      Wst[(c4 + 3) * 40 + k] = (__bf16)v.w;
    }
    __syncthreads();

    const int arow = w * 16 + l16;
    v16bf a = frag16(&As[arow * 40 + h * 8], &As[arow * 40 + 16 + h * 8]);
#pragma unroll
    for (int t = 0; t < 4; ++t) {
      int bcol = t * 16 + l16;
      v16bf b = frag16(&Wst[bcol * 40 + h * 8], &Wst[bcol * 40 + 16 + h * 8]);
      acc[t] = wmma_bf16(a, b, acc[t]);
    }
    __syncthreads();
  }

#pragma unroll
  for (int t = 0; t < 4; ++t) {
    int col = colbase + t * 16 + l16;
    float bv = bias[col];
#pragma unroll
    for (int r = 0; r < 8; ++r) {
      int grow = rowbase + w * 16 + r + h * 8;
      out[(size_t)grow * DIM + col] = acc[t][r] + bv;
    }
  }
}

// ---------------------------------------------------------------------------
extern "C" void kernel_launch(void* const* d_in, const int* in_sizes, int n_in,
                              void* d_out, int out_size, void* d_ws,
                              size_t ws_size, hipStream_t stream) {
  (void)in_sizes; (void)n_in; (void)out_size; (void)ws_size;
  const float* x   = (const float*)d_in[0];
  const float* Wq  = (const float*)d_in[1];
  const float* Wkv = (const float*)d_in[2];
  const float* Wo  = (const float*)d_in[3];
  const float* bo  = (const float*)d_in[4];
  const float* Wo0 = (const float*)d_in[5];
  const float* bo0 = (const float*)d_in[6];
  float* out = (float*)d_out;

  // Workspace (~52 MB total; fits in the 192 MB L2):
  __bf16* Qb  = (__bf16*)d_ws;                      // [32][4096][64] bf16
  __bf16* KVb = Qb + (size_t)32 * SEQN * DHEAD;     // [32][4096][64] bf16
  __bf16* A2  = KVb + (size_t)32 * SEQN * DHEAD;    // [8192][1024]  bf16
  float*  cosT = (float*)(A2 + (size_t)8192 * DIM); // [4096][64]    f32
  float*  sinT = cosT + (size_t)SEQN * DHEAD;       // [4096][64]    f32

  dim3 blk(256);
  rope_table_kernel<<<dim3(SEQN * DHEAD / 256), blk, 0, stream>>>(cosT, sinT);
  proj_rope_kernel<<<dim3(DIM / 64, 8192 / 128), blk, 0, stream>>>(
      x, Wq, cosT, sinT, Qb, 1);
  proj_rope_kernel<<<dim3(DIM / 64, 8192 / 128), blk, 0, stream>>>(
      x, Wkv, cosT, sinT, KVb, 0);
  attn_kernel<<<dim3(4, 8, 32), blk, 0, stream>>>(Qb, KVb, A2);
  out_proj_kernel<<<dim3(DIM / 64, 8192 / 128), blk, 0, stream>>>(
      A2, Wo, bo, Wo0, bo0, out);
}
